// AttentionLayer_49039936586174
// MI455X (gfx1250) — matrix operand order repaired and checked by reading, not compiled, not run on hardware
//
#include <hip/hip_runtime.h>
#include <hip/hip_bf16.h>

typedef __bf16 v16bf __attribute__((ext_vector_type(16)));
typedef float  v8f   __attribute__((ext_vector_type(8)));

#define B_DIM 64
#define S_DIM 2048
#define D_DIM 512
#define P_DIM 256
#define NROWS  (B_DIM * S_DIM)   // 131072 flat rows
#define NTILES (NROWS / 16)      // 8192 row tiles
#define KSTEPS (D_DIM / 32)      // 16 k-steps of 32

// ---------------------------------------------------------------------------
// Repack W [D,P] f32 -> bf16 B-fragments.
// Fragment layout used here, contiguous K order (B-matrix 32x16, wave32, 8 VGPRs):
//   lane = half*16 + n ; half selects K in [16*half, 16*half+16) ; N = n
//   element e (0..15) of the lane's v16bf holds K = kstep*32 + 16*half + e.
// Stored so each lane reads its fragment as one contiguous 32B vector.
// ---------------------------------------------------------------------------
__global__ __launch_bounds__(32)
void prep_wfrag(const float* __restrict__ W, __bf16* __restrict__ wfrag) {
    int nt   = blockIdx.x >> 4;   // N tile (0..15)
    int ks   = blockIdx.x & 15;   // K step (0..15)
    int lane = threadIdx.x;       // 0..31
    int half = lane >> 4;
    int col  = nt * 16 + (lane & 15);
    v16bf v;
#pragma unroll
    for (int e = 0; e < 16; ++e) {
        int K = ks * 32 + 16 * half + e;
        v[e] = (__bf16)W[K * P_DIM + col];
    }
    ((v16bf*)wfrag)[(nt * KSTEPS + ks) * 32 + lane] = v;
}

// ---------------------------------------------------------------------------
// Fused projection + tanh + u-dot -> per-row attention logits.
// One block = 8 waves handles 16 rows of flat [NROWS, D]; wave w computes
// P-column tiles {2w, 2w+1} with v_wmma_f32_16x16x32_bf16, then the epilogue
// reduces tanh(acc+b)*u over all 256 P columns into logits via shuffles +
// LDS float atomics. proj tensor never hits memory.
// ---------------------------------------------------------------------------
__global__ __launch_bounds__(256)
void proj_logits(const float* __restrict__ x, const __bf16* __restrict__ wfrag,
                 const float* __restrict__ bvec, const float* __restrict__ uvec,
                 float* __restrict__ logits) {
    __shared__ __bf16 ldsX[16 * D_DIM];   // 16 KB bf16 x-tile
    __shared__ float  rowsum[16];

    const int tid = threadIdx.x;
    const long tileBase = (long)blockIdx.x * (16 * D_DIM);

    if (tid < 16) rowsum[tid] = 0.0f;

    // cooperative load + f32->bf16 convert of the 16x512 tile (coalesced)
#pragma unroll
    for (int i = 0; i < 32; ++i) {
        int idx = i * 256 + tid;
        ldsX[idx] = (__bf16)x[tileBase + idx];
    }
    __syncthreads();

    const int lane = tid & 31;
    const int wave = tid >> 5;
    const int half = lane >> 4;
    const int mrow = lane & 15;

    v8f acc0 = {};
    v8f acc1 = {};
    const v16bf* wf = (const v16bf*)wfrag;
    const int nt0 = wave * 2, nt1 = wave * 2 + 1;
    const __bf16* xr = &ldsX[mrow * D_DIM];

    for (int ks = 0; ks < KSTEPS; ++ks) {
        // A fragment (16x32 bf16), contiguous K order matching B:
        //   e in [0,16): K = ks*32 + 16*half + e
        const int kb = ks * 32 + 16 * half;
        v16bf a;
#pragma unroll
        for (int e = 0; e < 8; ++e)  a[e] = xr[kb + e];
#pragma unroll
        for (int e = 8; e < 16; ++e) a[e] = xr[kb + e];

        v16bf b0 = wf[(nt0 * KSTEPS + ks) * 32 + lane];
        v16bf b1 = wf[(nt1 * KSTEPS + ks) * 32 + lane];

        acc0 = __builtin_amdgcn_wmma_f32_16x16x32_bf16(false, a, false, b0,
                                                       (short)0, acc0, false, false);
        acc1 = __builtin_amdgcn_wmma_f32_16x16x32_bf16(false, a, false, b1,
                                                       (short)0, acc1, false, false);
    }

    // Epilogue: lane holds column col, rows M = v + 8*half (C/D layout).
#pragma unroll
    for (int t = 0; t < 2; ++t) {
        v8f acc = t ? acc1 : acc0;
        int col = (wave * 2 + t) * 16 + (lane & 15);
        float bb = bvec[col];
        float uu = uvec[col];
#pragma unroll
        for (int v = 0; v < 8; ++v) {
            float p = tanhf(acc[v] + bb) * uu;
            // reduce over the 16 lanes of this half-wave (same row set)
            p += __shfl_xor(p, 8, 16);
            p += __shfl_xor(p, 4, 16);
            p += __shfl_xor(p, 2, 16);
            p += __shfl_xor(p, 1, 16);
            if ((lane & 15) == 0)
                atomicAdd(&rowsum[half * 8 + v], p);   // ds_add_f32
        }
    }
    __syncthreads();
    if (tid < 16) logits[(long)blockIdx.x * 16 + tid] = rowsum[tid];
}

// ---------------------------------------------------------------------------
// Softmax over S per batch; writes scores straight into d_out.
// ---------------------------------------------------------------------------
__global__ __launch_bounds__(256)
void softmax_scores(const float* __restrict__ logits, float* __restrict__ scores) {
    __shared__ float red[256];
    const int b = blockIdx.x, tid = threadIdx.x;
    const float* lb = logits + (long)b * S_DIM;

    float l[8];
    float m = -3.402823466e38f;
#pragma unroll
    for (int i = 0; i < 8; ++i) { l[i] = lb[tid + i * 256]; m = fmaxf(m, l[i]); }
    red[tid] = m; __syncthreads();
    for (int off = 128; off; off >>= 1) {
        if (tid < off) red[tid] = fmaxf(red[tid], red[tid + off]);
        __syncthreads();
    }
    m = red[0]; __syncthreads();

    float e[8], s = 0.0f;
#pragma unroll
    for (int i = 0; i < 8; ++i) { e[i] = __expf(l[i] - m); s += e[i]; }
    red[tid] = s; __syncthreads();
    for (int off = 128; off; off >>= 1) {
        if (tid < off) red[tid] += red[tid + off];
        __syncthreads();
    }
    float inv = 1.0f / red[0];

    float* sb = scores + (long)b * S_DIM;
#pragma unroll
    for (int i = 0; i < 8; ++i) sb[tid + i * 256] = e[i] * inv;
}

// ---------------------------------------------------------------------------
// context[b,d] = sum_s scores[b,s] * x[b,s,d]
// ---------------------------------------------------------------------------
__global__ __launch_bounds__(256)
void zero_ctx(float* __restrict__ ctx) {
    ctx[blockIdx.x * 256 + threadIdx.x] = 0.0f;
}

__global__ __launch_bounds__(256)
void context_accum(const float* __restrict__ x, const float* __restrict__ scores,
                   float* __restrict__ ctx) {
    __shared__ float sc[128];
    const int b = blockIdx.x, tid = threadIdx.x;
    const int s0 = blockIdx.y * 128;
    if (tid < 128) sc[tid] = scores[(long)b * S_DIM + s0 + tid];
    __syncthreads();

    const float* xb = x + ((long)b * S_DIM + s0) * D_DIM;
#pragma unroll
    for (int dd = 0; dd < 2; ++dd) {
        int d = tid + dd * 256;
        float a = 0.0f;
        for (int s = 0; s < 128; ++s)
            a += sc[s] * xb[(long)s * D_DIM + d];   // coalesced along d
        atomicAdd(&ctx[b * D_DIM + d], a);          // global_atomic_add_f32
    }
}

// ---------------------------------------------------------------------------
extern "C" void kernel_launch(void* const* d_in, const int* in_sizes, int n_in,
                              void* d_out, int out_size, void* d_ws, size_t ws_size,
                              hipStream_t stream) {
    const float* x  = (const float*)d_in[0];   // [B,S,D]
    const float* W  = (const float*)d_in[1];   // [D,P]
    const float* bb = (const float*)d_in[2];   // [P]
    const float* u  = (const float*)d_in[3];   // [P,1]

    float* ctx    = (float*)d_out;                     // [B,D]
    float* scores = (float*)d_out + B_DIM * D_DIM;     // [B,S,1]

    __bf16* wfrag = (__bf16*)d_ws;                                   // 256 KB
    float*  logits = (float*)((char*)d_ws + (size_t)(16 * 16 * 32 * 16) * 2); // 512 KB

    prep_wfrag<<<256, 32, 0, stream>>>(W, wfrag);
    proj_logits<<<NTILES, 256, 0, stream>>>(x, wfrag, bb, u, logits);
    softmax_scores<<<B_DIM, 256, 0, stream>>>(logits, scores);
    zero_ctx<<<(B_DIM * D_DIM) / 256, 256, 0, stream>>>(ctx);
    dim3 grid(B_DIM, S_DIM / 128);
    context_accum<<<grid, 256, 0, stream>>>(x, scores, ctx);
}
